// DENIS_JBF_57921928954562
// MI455X (gfx1250) — matrix operand order repaired and checked by reading, not compiled, not run on hardware
//
#include <hip/hip_runtime.h>

// ---------------------------------------------------------------------------
// Koopman autoencoder forward, MI455X (gfx1250, wave32, WMMA).
//   B=256, T=64, DIM=16, LDIM=256, H_AUX=N_AUX=128, DT=0.02
// Outputs: y (16384 x 272) then y_pred (16384 x 272), f32, concatenated.
// ---------------------------------------------------------------------------

#define DT_ 0.02f
#define B_    256
#define T_    64
#define DIM_  16
#define LDIM_ 256
#define HAUX_ 128
#define NAUX_ 128
#define ROWS_ (B_ * T_)          /* 16384 */
#define YLD_  (DIM_ + LDIM_)     /* 272   */

typedef __attribute__((ext_vector_type(16))) _Float16 v16h;
typedef __attribute__((ext_vector_type(8)))  _Float16 v8h;
typedef __attribute__((ext_vector_type(8)))  float    v8f;

// ---------------------------------------------------------------------------
// Prep: transpose+convert 256x256 f32 weight -> f16 n-major (Wt[n][k])
// ---------------------------------------------------------------------------
__global__ void k_transpose256(const float* __restrict__ W, _Float16* __restrict__ Wt) {
    int n = blockIdx.x;      // 0..255
    int k = threadIdx.x;     // 0..255
    Wt[n * 256 + k] = (_Float16)W[k * 256 + n];
}

// Wc (256x16) -> Wct (16 x 256 f16, n-major)
__global__ void k_transpose_wc(const float* __restrict__ Wc, _Float16* __restrict__ Wct) {
    int k = threadIdx.x;     // 0..255
    #pragma unroll
    for (int n = 0; n < 16; ++n)
        Wct[n * 256 + k] = (_Float16)Wc[k * 16 + n];
}

// y[:, :, 0:16] = x ; y_pred[:, 0, 0:16] = x[:, 0, :]
__global__ void k_scatter_x(const float* __restrict__ x,
                            float* __restrict__ y, float* __restrict__ ypred) {
    int idx = blockIdx.x * 256 + threadIdx.x;   // ROWS_*16 elements
    if (idx >= ROWS_ * DIM_) return;
    int row = idx >> 4;
    int d   = idx & 15;
    float v = x[idx];
    y[(size_t)row * YLD_ + d] = v;
    if ((row & (T_ - 1)) == 0)                  // t == 0 rows
        ypred[(size_t)row * YLD_ + d] = v;
}

// ---------------------------------------------------------------------------
// Aux net: per (b,n) compute scaled cos/sin of the 2x2 Koopman block.
// Block = one n (128 blocks), 256 threads = one b each. Weights in LDS.
// ---------------------------------------------------------------------------
__global__ void k_aux(const float* __restrict__ x,      // (B,T,DIM)
                      const float* __restrict__ aW1,    // (128,16,128)
                      const float* __restrict__ ab1,    // (128,128)
                      const float* __restrict__ aW2,    // (128,128,2)
                      float* __restrict__ cosv,         // (B,128)
                      float* __restrict__ sinv) {
    __shared__ float W1s[16 * 128];
    __shared__ float b1s[128];
    __shared__ float W2s[128 * 2];
    const int n = blockIdx.x;
    const int b = threadIdx.x;

    for (int i = threadIdx.x; i < 16 * 128; i += 256) W1s[i] = aW1[n * 2048 + i];
    if (threadIdx.x < 128) b1s[threadIdx.x] = ab1[n * 128 + threadIdx.x];
    W2s[threadIdx.x] = aW2[n * 256 + threadIdx.x];
    __syncthreads();

    float xr[16];
    #pragma unroll
    for (int d = 0; d < 16; ++d) xr[d] = x[(size_t)b * (T_ * DIM_) + d];  // x[b,0,d]

    float o0 = 0.f, o1 = 0.f;
    for (int j = 0; j < 128; ++j) {
        float h = b1s[j];
        #pragma unroll
        for (int d = 0; d < 16; ++d) h = fmaf(xr[d], W1s[d * 128 + j], h);
        h = fmaxf(h, 0.f);
        o0 = fmaf(h, W2s[j * 2 + 0], o0);
        o1 = fmaf(h, W2s[j * 2 + 1], o1);
    }
    float sc = __expf(o0 * DT_);
    float th = o1 * DT_;
    cosv[b * NAUX_ + n] = __cosf(th) * sc;
    sinv[b * NAUX_ + n] = __sinf(th) * sc;
}

// ---------------------------------------------------------------------------
// Encoder layer 1 (K=16, VALU): h1 = relu(x @ W1 + b1) -> f16
// Block = one row (16384 blocks), 256 threads = one output col each.
// ---------------------------------------------------------------------------
__global__ void k_layer1(const float* __restrict__ x, const float* __restrict__ W1,
                         const float* __restrict__ b1, _Float16* __restrict__ h1) {
    __shared__ float xr[16];
    const int r = blockIdx.x;
    const int j = threadIdx.x;
    if (j < 16) xr[j] = x[r * 16 + j];
    __syncthreads();
    float acc = b1[j];
    #pragma unroll
    for (int d = 0; d < 16; ++d) acc = fmaf(xr[d], W1[d * 256 + j], acc);
    h1[(size_t)r * 256 + j] = (_Float16)fmaxf(acc, 0.f);
}

// ---------------------------------------------------------------------------
// WMMA GEMM: C(ROWS_ x 256) = A(ROWS_ x 256, f16 row-major) @ B(256x256)
// with B pre-transposed to n-major f16 (Bt[n][k]).  v_wmma_f32_16x16x32_f16.
// Each wave owns one 16-row M-tile and 4 N-tiles (64 cols). 8 waves / block.
// Grid: 512 blocks x 256 threads  (1024 M-tiles * 4 N-quads = 4096 waves).
// ---------------------------------------------------------------------------
template <bool RELU, bool HAS_BIAS, bool OUT16, bool OUT32>
__global__ void k_wmma_gemm(const _Float16* __restrict__ A,
                            const _Float16* __restrict__ Bt,
                            const float* __restrict__ bias,
                            _Float16* __restrict__ out16,
                            float* out32 /* ld=YLD_, col offset 16 */) {
    const int lane = threadIdx.x & 31;
    const int wid  = threadIdx.x >> 5;
    const int gw   = blockIdx.x * 8 + wid;
    const int m0   = (gw >> 2) * 16;
    const int nq   = gw & 3;
    const int l    = lane & 15;
    const int hi   = lane >> 4;            // half-wave selector

    const _Float16* arow = A + (size_t)(m0 + l) * 256;

    v8f acc[4] = {};
    for (int k0 = 0; k0 < 256; k0 += 32) {
        // A fragment (16x32 f16): lanes 0-15 hold K 0..15, lanes 16-31 K 16..31
        v8h alo = *(const v8h*)(arow + k0 + hi * 16);
        v8h ahi = *(const v8h*)(arow + k0 + hi * 16 + 8);
        v16h a;
        #pragma unroll
        for (int i = 0; i < 8; ++i) { a[i] = alo[i]; a[i + 8] = ahi[i]; }
        #pragma unroll
        for (int nt = 0; nt < 4; ++nt) {
            // B fragment (32x16): N = n0+lane%16, lanes 0-15 K 0..15, lanes 16-31 K 16..31
            const int n = nq * 64 + nt * 16 + l;
            v16h bf = *(const v16h*)(Bt + (size_t)n * 256 + k0 + hi * 16);
            acc[nt] = __builtin_amdgcn_wmma_f32_16x16x32_f16(
                false, a, false, bf, (short)0, acc[nt], false, false);
        }
    }
    #pragma unroll
    for (int nt = 0; nt < 4; ++nt) {
        const int col = nq * 64 + nt * 16 + l;
        const float bv = HAS_BIAS ? bias[col] : 0.f;
        #pragma unroll
        for (int r = 0; r < 8; ++r) {
            const int row = m0 + r + hi * 8;   // C layout: VGPR r -> M=r (+8 for hi half)
            float v = acc[nt][r] + bv;
            if (RELU) v = fmaxf(v, 0.f);
            if (OUT16) out16[(size_t)row * 256 + col] = (_Float16)v;
            if (OUT32) out32[(size_t)row * YLD_ + DIM_ + col] = v;
        }
    }
}

// ---------------------------------------------------------------------------
// Koopman rollout: block per batch, thread per 2x2 block pair.
// y_pred[:, t, 16:272] = R(b)^t applied to y0 = y_enc[:,0,:]
// ---------------------------------------------------------------------------
__global__ void k_rollout(const float* __restrict__ y, const float* __restrict__ cosv,
                          const float* __restrict__ sinv, float* ypred) {
    const int b = blockIdx.x;     // 0..255
    const int n = threadIdx.x;    // 0..127
    const float c = cosv[b * NAUX_ + n];
    const float s = sinv[b * NAUX_ + n];
    const size_t base = (size_t)b * T_ * YLD_ + DIM_ + 2 * n;
    float a  = y[base + 0];
    float bb = y[base + 1];
    *(float2*)(ypred + base) = make_float2(a, bb);
    for (int t = 1; t < T_; ++t) {
        const float na = a * c - bb * s;
        const float nb = a * s + bb * c;
        a = na; bb = nb;
        *(float2*)(ypred + base + (size_t)t * YLD_) = make_float2(a, bb);
    }
}

// ---------------------------------------------------------------------------
// Decoder: x_pred[:,1:,:] = y_lat[:,1:,:] @ Wc via WMMA.
// M = 256*63 = 16128 rows (1008 tiles), N = 16, K = 256.
// A is f32 in y_pred latent columns, converted to f16 on load.
// Grid: 126 blocks x 256 threads (1008 waves).
// ---------------------------------------------------------------------------
__global__ void k_wc(float* ypred, const _Float16* __restrict__ Wct) {
    const int lane = threadIdx.x & 31;
    const int wid  = threadIdx.x >> 5;
    const int m0   = (blockIdx.x * 8 + wid) * 16;
    const int l    = lane & 15;
    const int hi   = lane >> 4;

    const int r  = m0 + l;           // 0..16127
    const int b  = r / 63;
    const int t  = 1 + r % 63;
    const float* rowp = ypred + ((size_t)b * T_ + t) * YLD_ + DIM_;

    v8f acc = {};
    for (int k0 = 0; k0 < 256; k0 += 32) {
        v8f f0 = *(const v8f*)(rowp + k0 + hi * 16);
        v8f f1 = *(const v8f*)(rowp + k0 + hi * 16 + 8);
        v16h a;
        #pragma unroll
        for (int i = 0; i < 8; ++i) {
            a[i]     = (_Float16)f0[i];
            a[i + 8] = (_Float16)f1[i];
        }
        v16h bf = *(const v16h*)(Wct + (size_t)l * 256 + k0 + hi * 16);
        acc = __builtin_amdgcn_wmma_f32_16x16x32_f16(
            false, a, false, bf, (short)0, acc, false, false);
    }
    #pragma unroll
    for (int rr = 0; rr < 8; ++rr) {
        const int row = m0 + rr + hi * 8;
        const int b2  = row / 63;
        const int t2  = 1 + row % 63;
        ypred[((size_t)b2 * T_ + t2) * YLD_ + l] = acc[rr];
    }
}

// ---------------------------------------------------------------------------
// Host launcher
// ---------------------------------------------------------------------------
extern "C" void kernel_launch(void* const* d_in, const int* in_sizes, int n_in,
                              void* d_out, int out_size, void* d_ws, size_t ws_size,
                              hipStream_t stream) {
    const float* x     = (const float*)d_in[0];
    const float* encW1 = (const float*)d_in[1];
    const float* encb1 = (const float*)d_in[2];
    const float* encW2 = (const float*)d_in[3];
    const float* encb2 = (const float*)d_in[4];
    const float* encW3 = (const float*)d_in[5];
    const float* auxW1 = (const float*)d_in[6];
    const float* auxb1 = (const float*)d_in[7];
    const float* auxW2 = (const float*)d_in[8];
    const float* Wc    = (const float*)d_in[9];

    float* y     = (float*)d_out;                     // (16384 x 272)
    float* ypred = y + (size_t)ROWS_ * YLD_;          // (16384 x 272)

    // Workspace layout (all offsets 32B-aligned); total ~17.3 MB
    char* ws = (char*)d_ws;
    _Float16* h1h  = (_Float16*)ws;  ws += (size_t)ROWS_ * 256 * sizeof(_Float16);
    _Float16* h2h  = (_Float16*)ws;  ws += (size_t)ROWS_ * 256 * sizeof(_Float16);
    _Float16* W2t  = (_Float16*)ws;  ws += 256 * 256 * sizeof(_Float16);
    _Float16* W3t  = (_Float16*)ws;  ws += 256 * 256 * sizeof(_Float16);
    _Float16* Wct  = (_Float16*)ws;  ws += 16 * 256 * sizeof(_Float16);
    float*    cosv = (float*)ws;     ws += (size_t)B_ * NAUX_ * sizeof(float);
    float*    sinv = (float*)ws;     ws += (size_t)B_ * NAUX_ * sizeof(float);

    // --- prep (independent) ---
    k_transpose256<<<256, 256, 0, stream>>>(encW2, W2t);
    k_transpose256<<<256, 256, 0, stream>>>(encW3, W3t);
    k_transpose_wc<<<1, 256, 0, stream>>>(Wc, Wct);
    k_scatter_x<<<(ROWS_ * DIM_ + 255) / 256, 256, 0, stream>>>(x, y, ypred);
    k_aux<<<NAUX_, 256, 0, stream>>>(x, auxW1, auxb1, auxW2, cosv, sinv);

    // --- encoder ---
    k_layer1<<<ROWS_, 256, 0, stream>>>(x, encW1, encb1, h1h);
    k_wmma_gemm<true,  true,  true,  false><<<512, 256, 0, stream>>>(h1h, W2t, encb2, h2h, nullptr);
    k_wmma_gemm<false, false, false, true ><<<512, 256, 0, stream>>>(h2h, W3t, nullptr, nullptr, y);

    // --- rollout + decode ---
    k_rollout<<<B_, NAUX_, 0, stream>>>(y, cosv, sinv, ypred);
    k_wc<<<126, 256, 0, stream>>>(ypred, Wct);
}
